// EncoderCoreDecoder_23587960390348
// MI455X (gfx1250) — compile-verified, matrix operand order checked by reading, not run on hardware
//
#include <hip/hip_runtime.h>
#include <hip/hip_bf16.h>

typedef unsigned short u16;
typedef __attribute__((ext_vector_type(16))) __bf16 v16bf;
typedef __attribute__((ext_vector_type(8)))  float  v8f;
typedef __attribute__((ext_vector_type(4)))  unsigned int u32x4;
typedef __attribute__((ext_vector_type(8)))  int  i32x8;
typedef __attribute__((ext_vector_type(4)))  int  i32x4;

#define B_    4
#define V_    1024
#define S_    32
#define EMB_  128
#define HID_  256
#define EROWS (B_*V_*S_)   /* 131072 */
#define VROWS (B_*V_)      /* 4096   */

#if __has_builtin(__builtin_amdgcn_tensor_load_to_lds) && __has_builtin(__builtin_amdgcn_s_wait_tensorcnt)
#define USE_TDM 1
#else
#define USE_TDM 0
#endif

__device__ __forceinline__ u16 f2bf(float x){
  unsigned u = __float_as_uint(x);
  unsigned r = u + 0x7FFFu + ((u >> 16) & 1u);
  return (u16)(r >> 16);
}
__device__ __forceinline__ float bf2f(u16 h){
  return __uint_as_float(((unsigned)h) << 16);
}

#if USE_TDM
// ---------------------------------------------------------------------------
// Issue a TDM 2D tile load: rows x 32 bf16 elements from global (row stride K
// elements) into LDS at ldsAddr, padding each 64B row with 16B in LDS so the
// destination row stride is 40 halves (= our LDS tile layout).
// D# packing per CDNA5 ISA 8.3/8.4: group0 {count=1, lds_addr, global_addr,
// type=2}; group1 {data_size=2B, pad_enable, pad_interval=16dw, pad_amount=4dw,
// tensor_dim0=K, tensor_dim1=rows, tile_dim0=32, tile_dim1=128,
// tensor_dim0_stride=K}; groups 2/3 zero (2D tensor, tile_dim2=0).
// This toolchain exposes the 6-arg builtin (extra int32x8 group, zero-filled).
// ---------------------------------------------------------------------------
__device__ __forceinline__ void tdm_load_tile(unsigned ldsAddr, const u16* gsrc,
                                              unsigned K, unsigned rows)
{
  unsigned long long ga = (unsigned long long)(size_t)gsrc;
  u32x4 g0;
  g0.x = 1u;                                            // count=1 (valid user desc)
  g0.y = ldsAddr;                                       // lds_addr bytes
  g0.z = (unsigned)(ga & 0xFFFFFFFFu);                  // global_addr[31:0]
  g0.w = (unsigned)((ga >> 32) & 0x01FFFFFFu) | (2u << 30);  // addr[56:32] | type=2
  i32x8 g1;
  g1[0] = (int)((1u << 16) | (1u << 20) | (3u << 22) | (3u << 25)); // 2B elems, pad 16dw->+4dw
  g1[1] = (int)((K & 0xFFFFu) << 16);                   // tensor_dim0[15:0] @ bits 63:48
  g1[2] = (int)(((K >> 16) & 0xFFFFu) | ((rows & 0xFFFFu) << 16)); // dim0 hi | dim1 lo
  g1[3] = (int)(((rows >> 16) & 0xFFFFu) | (32u << 16)); // dim1 hi | tile_dim0=32
  g1[4] = (int)128;                                     // tile_dim1=128, tile_dim2=0
  g1[5] = (int)K;                                       // tensor_dim0_stride[31:0]
  g1[6] = 0;                                            // stride hi | dim1_stride lo
  g1[7] = 0;
  i32x4 gz4 = (i32x4){0, 0, 0, 0};
  i32x8 gz8 = (i32x8){0, 0, 0, 0, 0, 0, 0, 0};
  __builtin_amdgcn_tensor_load_to_lds(g0, g1, gz4, gz4, gz8, 0);
}
#endif

// ---------------------------------------------------------------------------
// Generic bf16 GEMM:  C[M,N] = act(A[M,K] @ W[K,N] + bias[N])
// A row-major bf16 [M,K]; WT is the PRE-TRANSPOSED weight, row-major [N,K];
// bias fp32; C bf16. Requires K % 32 == 0, N % 128 == 0.
// Block = 256 threads (8 waves). Block tile 128x128, K-step 32, double-
// buffered LDS fed by the Tensor Data Mover (wave 0 issues tensor_load_to_lds
// for A and W tiles; TENSORcnt + workgroup barrier synchronize); each wave
// computes 16(M) x 128(N) with 8 back-to-back V_WMMA_F32_16X16X32_BF16.
// ---------------------------------------------------------------------------
__global__ __launch_bounds__(256)
void gemm_bf16_kernel(const u16* __restrict__ A, const u16* __restrict__ WT,
                      const float* __restrict__ bias, u16* __restrict__ C,
                      int M, int N, int K, int relu)
{
  // stride 40 halves (80B) keeps 8-half chunks 16B-aligned and staggers banks
  __shared__ __align__(16) u16 As[2][128][40];  // [buf][m][k]
  __shared__ __align__(16) u16 Ws[2][128][40];  // [buf][n][k]

  const int bm   = blockIdx.x * 128;
  const int bn   = blockIdx.y * 128;
  const int tid  = threadIdx.x;
  const int wave = tid >> 5;
  const int lane = tid & 31;
  const int m16  = lane & 15;   // row/col within 16-block
  const int hh   = lane >> 4;   // half-wave select

  v8f acc[8];
#pragma unroll
  for (int j = 0; j < 8; ++j) acc[j] = (v8f){};

  union FragBF { v16bf v; u32x4 u[2]; };

#if USE_TDM
  const u16* aTile = A  + (size_t)bm * K;
  const u16* wTile = WT + (size_t)bn * K;
  const unsigned aRows = (unsigned)(M - bm);   // OOB rows read as zero by TDM
  const unsigned wRows = (unsigned)(N - bn);
  const unsigned ldsA0 = (unsigned)(size_t)&As[0][0][0];
  const unsigned ldsA1 = (unsigned)(size_t)&As[1][0][0];
  const unsigned ldsW0 = (unsigned)(size_t)&Ws[0][0][0];
  const unsigned ldsW1 = (unsigned)(size_t)&Ws[1][0][0];

  if (wave == 0) {
    tdm_load_tile(ldsA0, aTile, (unsigned)K, aRows);
    tdm_load_tile(ldsW0, wTile, (unsigned)K, wRows);
    __builtin_amdgcn_s_wait_tensorcnt(0);
  }
  __syncthreads();

  int p = 0;
  for (int k0 = 0; k0 < K; k0 += 32) {
    const int  kn   = k0 + 32;
    const bool more = kn < K;

    // DMA the next tile into the back buffer while we compute on this one
    if (more && wave == 0) {
      tdm_load_tile(p ? ldsA0 : ldsA1, aTile + kn, (unsigned)K, aRows);
      tdm_load_tile(p ? ldsW0 : ldsW1, wTile + kn, (unsigned)K, wRows);
    }
#else
  // staging role: thread handles one row, one 16-half chunk
  const int lr = tid >> 1;           // 0..127
  const int lc = (tid & 1) * 16;     // 0 or 16
  const u16* aRow = A  + (size_t)(bm + lr) * K + lc;
  const u16* wRow = WT + (size_t)(bn + lr) * K + lc;
  const bool aOk  = (bm + lr) < M;
  {
    u32x4 a0 = (u32x4){}, a1 = (u32x4){};
    if (aOk) { a0 = *(const u32x4*)(aRow); a1 = *(const u32x4*)(aRow + 8); }
    u32x4 w0 = *(const u32x4*)(wRow);
    u32x4 w1 = *(const u32x4*)(wRow + 8);
    *(u32x4*)&As[0][lr][lc]     = a0;
    *(u32x4*)&As[0][lr][lc + 8] = a1;
    *(u32x4*)&Ws[0][lr][lc]     = w0;
    *(u32x4*)&Ws[0][lr][lc + 8] = w1;
  }
  __syncthreads();

  int p = 0;
  for (int k0 = 0; k0 < K; k0 += 32) {
    const int  kn   = k0 + 32;
    const bool more = kn < K;

    u32x4 na0 = (u32x4){}, na1 = (u32x4){}, nw0 = (u32x4){}, nw1 = (u32x4){};
    if (more) {
      if (aOk) { na0 = *(const u32x4*)(aRow + kn); na1 = *(const u32x4*)(aRow + kn + 8); }
      nw0 = *(const u32x4*)(wRow + kn);
      nw1 = *(const u32x4*)(wRow + kn + 8);
      if (kn + 32 < K) {
        if (aOk) __builtin_prefetch(aRow + kn + 32, 0, 1);
        __builtin_prefetch(wRow + kn + 32, 0, 1);
      }
    }
#endif

    // ---- fragments from buffer p: issue ALL ds loads, then all WMMAs ----
    FragBF af;
    af.u[0] = *(const u32x4*)&As[p][wave * 16 + m16][hh * 8];
    af.u[1] = *(const u32x4*)&As[p][wave * 16 + m16][16 + hh * 8];

    FragBF bfr[8];
#pragma unroll
    for (int j = 0; j < 8; ++j) {
      const int n = j * 16 + m16;
      bfr[j].u[0] = *(const u32x4*)&Ws[p][n][hh * 8];
      bfr[j].u[1] = *(const u32x4*)&Ws[p][n][16 + hh * 8];
    }

    __builtin_amdgcn_sched_barrier(0);   // keep ds_loads ahead of the WMMA burst

#pragma unroll
    for (int j = 0; j < 8; ++j) {
      acc[j] = __builtin_amdgcn_wmma_f32_16x16x32_bf16(
          false, af.v, false, bfr[j].v, (short)0, acc[j], false, false);
    }

#if USE_TDM
    if (more) {
      if (wave == 0) __builtin_amdgcn_s_wait_tensorcnt(0);
      __syncthreads();                   // publish back buffer to all waves
      p ^= 1;
    }
  }
#else
    if (more) {
      const int q = p ^ 1;
      *(u32x4*)&As[q][lr][lc]     = na0;
      *(u32x4*)&As[q][lr][lc + 8] = na1;
      *(u32x4*)&Ws[q][lr][lc]     = nw0;
      *(u32x4*)&Ws[q][lr][lc + 8] = nw1;
      __syncthreads();
      p = q;
    }
  }
#endif

  // ---- epilogue: bias + optional relu, bf16 store ----
#pragma unroll
  for (int j = 0; j < 8; ++j) {
#pragma unroll
    for (int r = 0; r < 8; ++r) {
      const int ml = wave * 16 + hh * 8 + r;
      const int gm = bm + ml;
      const int gn = bn + j * 16 + m16;
      if (gm < M) {
        float x = acc[j][r] + bias[gn];
        if (relu) x = fmaxf(x, 0.f);
        C[(size_t)gm * N + gn] = f2bf(x);
      }
    }
  }
}

// ---------------------------------------------------------------------------
// fp32 weights [K,N] -> bf16 TRANSPOSED [N,Kpad] (zero for k >= K)
// ---------------------------------------------------------------------------
__global__ void cvt_w_transpose(const float* __restrict__ src, u16* __restrict__ dst,
                                int K, int Kpad, int N)
{
  int i = blockIdx.x * blockDim.x + threadIdx.x;
  int total = N * Kpad;
  if (i >= total) return;
  int n = i / Kpad, k = i % Kpad;
  float v = (k < K) ? src[(size_t)k * N + n] : 0.f;
  dst[i] = f2bf(v);
}

// fp32 [M,K] activations -> bf16 [M,Kpad] (zero cols for k >= K)
__global__ void cvt_pad_cols(const float* __restrict__ src, u16* __restrict__ dst,
                             int M, int K, int Kpad)
{
  int i = blockIdx.x * blockDim.x + threadIdx.x;
  int total = M * Kpad;
  if (i >= total) return;
  int m = i / Kpad, k = i % Kpad;
  float v = (k < K) ? src[(size_t)m * K + k] : 0.f;
  dst[i] = f2bf(v);
}

__global__ void zero_f32(float* p, int n){
  int i = blockIdx.x * blockDim.x + threadIdx.x;
  if (i < n) p[i] = 0.f;
}

// per-(b,v) mask counts + per-graph edge counts (invariant across iterations)
__global__ void precompute_counts(const int* __restrict__ incoming,
                                  int* __restrict__ numSend, float* __restrict__ numEdges)
{
  int bv = blockIdx.x * blockDim.x + threadIdx.x;
  if (bv >= VROWS) return;
  int cnt = 0;
#pragma unroll
  for (int s = 0; s < S_; ++s) cnt += (incoming[bv * S_ + s] > 0) ? 1 : 0;
  numSend[bv] = cnt;
  atomicAdd(&numEdges[bv >> 10], (float)cnt);
}

// e_net input: [emb_E | recV | sendV | rC] -> eIn [EROWS, 512] bf16
__global__ void build_e_in(const u16* __restrict__ embE, const u16* __restrict__ embV,
                           const u16* __restrict__ embC, const int* __restrict__ incoming,
                           u16* __restrict__ eIn)
{
  int idx = blockIdx.x * blockDim.x + threadIdx.x;
  int total = EROWS * 64;               // 64 chunks of 8 halves per row
  if (idx >= total) return;
  int chunk = idx & 63;
  int row   = idx >> 6;
  int v = (row >> 5) & (V_ - 1);
  int b = row >> 15;
  int col = chunk * 8;
  int inc = incoming[row];
  u32x4 out = (u32x4){};
  if (col < 128) {
    out = *(const u32x4*)(embE + (size_t)row * EMB_ + col);
  } else if (col < 256) {
    if (inc > 0) out = *(const u32x4*)(embV + ((size_t)(b * V_ + v)) * EMB_ + (col - 128));
  } else if (col < 384) {
    if (inc > 0) out = *(const u32x4*)(embV + ((size_t)(b * V_ + (inc - 1))) * EMB_ + (col - 256));
  } else {
    out = *(const u32x4*)(embC + (size_t)b * EMB_ + (col - 384));
  }
  *(u32x4*)(eIn + (size_t)row * 512 + col) = out;
}

// masked edge sum over S; writes v_net input [avgE | emb_V | emb_C] and accumulates
// per-graph edge-feature sums. One block per (b,v), 128 threads (one per channel).
__global__ void edge_agg(const u16* __restrict__ embE, const u16* __restrict__ embV,
                         const u16* __restrict__ embC, const int* __restrict__ incoming,
                         const int* __restrict__ numSend,
                         u16* __restrict__ vIn, float* __restrict__ eAgg)
{
  int bv = blockIdx.x;
  int n  = threadIdx.x;
  int b  = bv >> 10;
  float sum = 0.f;
#pragma unroll 4
  for (int s = 0; s < S_; ++s) {
    if (incoming[bv * S_ + s] > 0)
      sum += bf2f(embE[((size_t)(bv * S_ + s)) * EMB_ + n]);
  }
  int ns = numSend[bv];
  float avg = sum / (float)(ns > 1 ? ns : 1);
  vIn[(size_t)bv * 384 + n]       = f2bf(avg);
  vIn[(size_t)bv * 384 + 128 + n] = embV[(size_t)bv * EMB_ + n];
  vIn[(size_t)bv * 384 + 256 + n] = embC[(size_t)b * EMB_ + n];
  atomicAdd(&eAgg[b * EMB_ + n], sum);
}

// c_net input: [global_agg_e | global_agg_v | emb_C] ; one block per graph.
__global__ void build_c_in(const u16* __restrict__ embV, const u16* __restrict__ embC,
                           const float* __restrict__ eAgg, const float* __restrict__ numEdges,
                           u16* __restrict__ cIn)
{
  int b = blockIdx.x, n = threadIdx.x;
  float gv = 0.f;
  for (int v = 0; v < V_; ++v) gv += bf2f(embV[((size_t)(b * V_ + v)) * EMB_ + n]);
  gv *= (1.f / (float)V_);
  float ge = eAgg[b * EMB_ + n] / fmaxf(numEdges[b], 1.f);
  cIn[b * 384 + n]       = f2bf(ge);
  cIn[b * 384 + 128 + n] = f2bf(gv);
  cIn[b * 384 + 256 + n] = embC[b * EMB_ + n];
}

// final decoder layer (hidden 256 -> 2), fp32 weights/out
__global__ void dec_final(const u16* __restrict__ h, const float* __restrict__ W3,
                          const float* __restrict__ b3, float* __restrict__ out, int M)
{
  int m = blockIdx.x * blockDim.x + threadIdx.x;
  if (m >= M) return;
  float z0 = b3[0], z1 = b3[1];
  for (int k = 0; k < HID_; ++k) {
    float hv = bf2f(h[(size_t)m * HID_ + k]);
    z0 += hv * W3[k * 2 + 0];
    z1 += hv * W3[k * 2 + 1];
  }
  out[(size_t)m * 2 + 0] = z0;
  out[(size_t)m * 2 + 1] = z1;
}

// dec_e final layer + log_softmax + mask
__global__ void dec_e_final(const u16* __restrict__ h, const float* __restrict__ W3,
                            const float* __restrict__ b3, const int* __restrict__ incoming,
                            float* __restrict__ out)
{
  int m = blockIdx.x * blockDim.x + threadIdx.x;
  if (m >= EROWS) return;
  float z0 = b3[0], z1 = b3[1];
  for (int k = 0; k < HID_; ++k) {
    float hv = bf2f(h[(size_t)m * HID_ + k]);
    z0 += hv * W3[k * 2 + 0];
    z1 += hv * W3[k * 2 + 1];
  }
  float o0 = 0.f, o1 = 0.f;
  if (incoming[m] > 0) {
    float mx  = fmaxf(z0, z1);
    float lse = mx + __logf(__expf(z0 - mx) + __expf(z1 - mx));
    o0 = z0 - lse; o1 = z1 - lse;
  }
  out[(size_t)m * 2 + 0] = o0;
  out[(size_t)m * 2 + 1] = o1;
}

// ---------------------------------------------------------------------------
extern "C" void kernel_launch(void* const* d_in, const int* in_sizes, int n_in,
                              void* d_out, int out_size, void* d_ws, size_t ws_size,
                              hipStream_t stream)
{
  (void)in_sizes; (void)n_in; (void)out_size; (void)ws_size;

  const float* vertex   = (const float*)d_in[0];
  const float* edge     = (const float*)d_in[1];
  const float* context  = (const float*)d_in[2];
  const int*   incoming = (const int*)  d_in[3];
  // nets: 6 leaves each (W1,b1,W2,b2,W3,b3)
  const int ENC_V = 4, ENC_E = 10, ENC_C = 16, VNET = 22, ENET = 28, CNET = 34,
            DECV = 40, DECE = 46, DECC = 52;
  auto F = [&](int i) { return (const float*)d_in[i]; };

  // ---- workspace bump allocator (256B aligned) ----
  char* ws = (char*)d_ws;
  size_t off = 0;
  auto allocB = [&](size_t bytes) -> void* {
    void* p = (void*)(ws + off);
    off = (off + bytes + 255) & ~(size_t)255;
    return p;
  };
  u16* bufA = (u16*)allocB((size_t)EROWS * 512 * 2);  // e-input / hidden reuse
  u16* bufH = (u16*)allocB((size_t)EROWS * HID_ * 2); // hidden
  u16* embE = (u16*)allocB((size_t)EROWS * EMB_ * 2);
  u16* embV = (u16*)allocB((size_t)VROWS * EMB_ * 2);
  u16* embC = (u16*)allocB((size_t)B_ * EMB_ * 2);
  u16* vIn  = (u16*)allocB((size_t)VROWS * 384 * 2);
  u16* cIn  = (u16*)allocB((size_t)B_ * 384 * 2);
  u16* hV   = (u16*)allocB((size_t)VROWS * HID_ * 2);
  u16* hV2  = (u16*)allocB((size_t)VROWS * HID_ * 2);
  u16* hC   = (u16*)allocB((size_t)B_ * HID_ * 2);
  u16* hC2  = (u16*)allocB((size_t)B_ * HID_ * 2);
  u16* Av   = (u16*)allocB((size_t)VROWS * 32 * 2);
  u16* Ac   = (u16*)allocB((size_t)B_ * 32 * 2);
  float* eAgg     = (float*)allocB((size_t)B_ * EMB_ * 4);
  float* numEdges = (float*)allocB((size_t)B_ * 4);
  int*   numSend  = (int*)  allocB((size_t)VROWS * 4);
  // bf16 TRANSPOSED weights [N, Kpad]
  u16* wEncV1 = (u16*)allocB(256 * 32 * 2);
  u16* wEncV2 = (u16*)allocB(256 * 256 * 2);
  u16* wEncV3 = (u16*)allocB(128 * 256 * 2);
  u16* wEncE1 = (u16*)allocB(256 * 32 * 2);
  u16* wEncE2 = (u16*)allocB(256 * 256 * 2);
  u16* wEncE3 = (u16*)allocB(128 * 256 * 2);
  u16* wEncC1 = (u16*)allocB(256 * 32 * 2);
  u16* wEncC2 = (u16*)allocB(256 * 256 * 2);
  u16* wEncC3 = (u16*)allocB(128 * 256 * 2);
  u16* wV1 = (u16*)allocB(256 * 384 * 2);
  u16* wV2 = (u16*)allocB(256 * 256 * 2);
  u16* wV3 = (u16*)allocB(128 * 256 * 2);
  u16* wE1 = (u16*)allocB(256 * 512 * 2);
  u16* wE2 = (u16*)allocB(256 * 256 * 2);
  u16* wE3 = (u16*)allocB(128 * 256 * 2);
  u16* wC1 = (u16*)allocB(256 * 384 * 2);
  u16* wC2 = (u16*)allocB(256 * 256 * 2);
  u16* wC3 = (u16*)allocB(128 * 256 * 2);
  u16* wDV1 = (u16*)allocB(256 * 128 * 2);
  u16* wDV2 = (u16*)allocB(256 * 256 * 2);
  u16* wDE1 = (u16*)allocB(256 * 128 * 2);
  u16* wDE2 = (u16*)allocB(256 * 256 * 2);
  u16* wDC1 = (u16*)allocB(256 * 128 * 2);
  u16* wDC2 = (u16*)allocB(256 * 256 * 2);

  auto gemm = [&](const u16* A, const u16* WT, const float* bias, u16* C,
                  int M, int N, int K, int relu) {
    dim3 g((M + 127) / 128, N / 128);
    gemm_bf16_kernel<<<g, 256, 0, stream>>>(A, WT, bias, C, M, N, K, relu);
  };
  auto cvtW = [&](const float* src, u16* dst, int K, int Kpad, int N) {
    int total = N * Kpad;
    cvt_w_transpose<<<(total + 255) / 256, 256, 0, stream>>>(src, dst, K, Kpad, N);
  };
  auto cvtA = [&](const float* src, u16* dst, int M, int K, int Kpad) {
    int total = M * Kpad;
    cvt_pad_cols<<<(total + 255) / 256, 256, 0, stream>>>(src, dst, M, K, Kpad);
  };

  // ---- weight conversions (bf16 transposed, K padded to 32 for 8-wide encoders) ----
  cvtW(F(ENC_V + 0), wEncV1, 8, 32, 256);  cvtW(F(ENC_V + 2), wEncV2, 256, 256, 256);
  cvtW(F(ENC_V + 4), wEncV3, 256, 256, 128);
  cvtW(F(ENC_E + 0), wEncE1, 8, 32, 256);  cvtW(F(ENC_E + 2), wEncE2, 256, 256, 256);
  cvtW(F(ENC_E + 4), wEncE3, 256, 256, 128);
  cvtW(F(ENC_C + 0), wEncC1, 8, 32, 256);  cvtW(F(ENC_C + 2), wEncC2, 256, 256, 256);
  cvtW(F(ENC_C + 4), wEncC3, 256, 256, 128);
  cvtW(F(VNET + 0), wV1, 384, 384, 256);   cvtW(F(VNET + 2), wV2, 256, 256, 256);
  cvtW(F(VNET + 4), wV3, 256, 256, 128);
  cvtW(F(ENET + 0), wE1, 512, 512, 256);   cvtW(F(ENET + 2), wE2, 256, 256, 256);
  cvtW(F(ENET + 4), wE3, 256, 256, 128);
  cvtW(F(CNET + 0), wC1, 384, 384, 256);   cvtW(F(CNET + 2), wC2, 256, 256, 256);
  cvtW(F(CNET + 4), wC3, 256, 256, 128);
  cvtW(F(DECV + 0), wDV1, 128, 128, 256);  cvtW(F(DECV + 2), wDV2, 256, 256, 256);
  cvtW(F(DECE + 0), wDE1, 128, 128, 256);  cvtW(F(DECE + 2), wDE2, 256, 256, 256);
  cvtW(F(DECC + 0), wDC1, 128, 128, 256);  cvtW(F(DECC + 2), wDC2, 256, 256, 256);

  // ---- invariant counts ----
  zero_f32<<<1, 32, 0, stream>>>(numEdges, B_);
  precompute_counts<<<(VROWS + 255) / 256, 256, 0, stream>>>(incoming, numSend, numEdges);

  // ---- encoders ----
  cvtA(vertex, Av, VROWS, 8, 32);
  gemm(Av, wEncV1, F(ENC_V + 1), hV,  VROWS, 256, 32, 1);
  gemm(hV, wEncV2, F(ENC_V + 3), hV2, VROWS, 256, 256, 1);
  gemm(hV2, wEncV3, F(ENC_V + 5), embV, VROWS, 128, 256, 0);

  cvtA(edge, bufA, EROWS, 8, 32);
  gemm(bufA, wEncE1, F(ENC_E + 1), bufH, EROWS, 256, 32, 1);
  gemm(bufH, wEncE2, F(ENC_E + 3), bufA, EROWS, 256, 256, 1);
  gemm(bufA, wEncE3, F(ENC_E + 5), embE, EROWS, 128, 256, 0);

  cvtA(context, Ac, B_, 8, 32);
  gemm(Ac, wEncC1, F(ENC_C + 1), hC,  B_, 256, 32, 1);
  gemm(hC, wEncC2, F(ENC_C + 3), hC2, B_, 256, 256, 1);
  gemm(hC2, wEncC3, F(ENC_C + 5), embC, B_, 128, 256, 0);

  float* outV = (float*)d_out;
  float* outE = outV + (size_t)3 * VROWS * 2;
  float* outC = outE + (size_t)3 * EROWS * 2;

  // ---- message-passing iterations ----
  for (int t = 0; t < 3; ++t) {
    // e_net
    build_e_in<<<(EROWS * 64 + 255) / 256, 256, 0, stream>>>(embE, embV, embC, incoming, bufA);
    gemm(bufA, wE1, F(ENET + 1), bufH, EROWS, 256, 512, 1);
    gemm(bufH, wE2, F(ENET + 3), bufA, EROWS, 256, 256, 1);
    gemm(bufA, wE3, F(ENET + 5), embE, EROWS, 128, 256, 0);

    // edge aggregation + v_net
    zero_f32<<<2, 256, 0, stream>>>(eAgg, B_ * EMB_);
    edge_agg<<<VROWS, 128, 0, stream>>>(embE, embV, embC, incoming, numSend, vIn, eAgg);
    gemm(vIn, wV1, F(VNET + 1), hV,  VROWS, 256, 384, 1);
    gemm(hV, wV2, F(VNET + 3), hV2, VROWS, 256, 256, 1);
    gemm(hV2, wV3, F(VNET + 5), embV, VROWS, 128, 256, 0);

    // global aggregation + c_net
    build_c_in<<<B_, 128, 0, stream>>>(embV, embC, eAgg, numEdges, cIn);
    gemm(cIn, wC1, F(CNET + 1), hC,  B_, 256, 384, 1);
    gemm(hC, wC2, F(CNET + 3), hC2, B_, 256, 256, 1);
    gemm(hC2, wC3, F(CNET + 5), embC, B_, 128, 256, 0);

    // decoders
    gemm(embV, wDV1, F(DECV + 1), hV,  VROWS, 256, 128, 1);
    gemm(hV, wDV2, F(DECV + 3), hV2, VROWS, 256, 256, 1);
    dec_final<<<(VROWS + 255) / 256, 256, 0, stream>>>(hV2, F(DECV + 4), F(DECV + 5),
                                                       outV + (size_t)t * VROWS * 2, VROWS);

    gemm(embE, wDE1, F(DECE + 1), bufH, EROWS, 256, 128, 1);
    gemm(bufH, wDE2, F(DECE + 3), bufA, EROWS, 256, 256, 1);
    dec_e_final<<<(EROWS + 255) / 256, 256, 0, stream>>>(bufA, F(DECE + 4), F(DECE + 5),
                                                         incoming, outE + (size_t)t * EROWS * 2);

    gemm(embC, wDC1, F(DECC + 1), hC,  B_, 256, 128, 1);
    gemm(hC, wDC2, F(DECC + 3), hC2, B_, 256, 256, 1);
    dec_final<<<1, 256, 0, stream>>>(hC2, F(DECC + 4), F(DECC + 5),
                                     outC + (size_t)t * B_ * 2, B_);
  }
}
